// Model_5033701671378
// MI455X (gfx1250) — matrix operand order repaired and checked by reading, not compiled, not run on hardware
//
#include <hip/hip_runtime.h>
#include <hip/hip_bf16.h>

// ---------------------------------------------------------------------------
// Bidirectional 4-layer SRU (H=512, B=8, L=2048) + classifier for gfx1250.
// GEMMs run as bf16 WMMA (v_wmma_f32_16x16x32_bf16), recurrence in f32.
// ---------------------------------------------------------------------------

typedef __bf16 bf16_t;
typedef __attribute__((ext_vector_type(16))) __bf16 v16bf;
typedef __attribute__((ext_vector_type(8)))  float  v8f;

#define H     512
#define BDIM  8
#define LSEQ  2048
#define MROWS (LSEQ * BDIM)   // 16384
#define KIN   1024            // 2*H (input width, layers 1..3)
#define NCOLS 3072            // 2*3*H (output width, layers 1..3)

__device__ __forceinline__ float sigmoidf_(float x) {
    return 1.0f / (1.0f + __expf(-x));
}

// ---------------------------------------------------------------------------
// f32 -> bf16 conversion (weights / staging)
// ---------------------------------------------------------------------------
__global__ void f32_to_bf16_kernel(const float* __restrict__ src,
                                   bf16_t* __restrict__ dst, int n) {
    int i = blockIdx.x * blockDim.x + threadIdx.x;
    if (i < n) dst[i] = (bf16_t)src[i];
}

// ---------------------------------------------------------------------------
// Layer 0: n_in = 1, k = 4.  U[m, dir, j, h] = dx[m] * W0[0, (dir*4+j)*H + h].
// dX = X - roll(X,1) with first step zeroed, already folded in here.
// Writes the channel-interleaved float4 layout Uperm[(m*2+dir)*H + h].
// ---------------------------------------------------------------------------
__global__ void layer0_u_kernel(const float* __restrict__ X,
                                const float* __restrict__ W0,
                                float4* __restrict__ Uperm) {
    int tid = blockIdx.x * blockDim.x + threadIdx.x;   // 0 .. MROWS*2*H-1
    int h   = tid & (H - 1);
    int dir = (tid >> 9) & 1;
    int m   = tid >> 10;
    int l   = m >> 3;
    int b   = m & 7;
    float dx = 0.0f;
    if (l > 0) dx = X[b * LSEQ + l] - X[b * LSEQ + l - 1];
    float4 u;
    u.x = dx * W0[(dir * 4 + 0) * H + h];
    u.y = dx * W0[(dir * 4 + 1) * H + h];
    u.z = dx * W0[(dir * 4 + 2) * H + h];
    u.w = dx * W0[(dir * 4 + 3) * H + h];   // k==4: residual = u3
    Uperm[(m * 2 + dir) * H + h] = u;
}

// ---------------------------------------------------------------------------
// bf16 WMMA GEMM for layers 1..3:  U[M=16384, N=3072] = Xb[M,K=1024] @ Wb[K,N]
// Workgroup: 256 threads (8 waves), tile 128(M) x 64(N), K-step 32.
// Wave w computes rows [tileM + 16w, +16) against 4 N-subtiles of 16.
// Epilogue scatters into Uperm[m][dir][h][4]; slot 3 = residual from Xb.
// ---------------------------------------------------------------------------
__global__ __launch_bounds__(256)
void sru_gemm_kernel(const bf16_t* __restrict__ Xb,
                     const bf16_t* __restrict__ Wb,
                     float* __restrict__ Uperm) {
    __shared__ __align__(16) bf16_t As[128][40];  // 128x32 A tile (+pad)
    __shared__ __align__(16) bf16_t Bs[64][40];   // 64(N) x 32(K) B^T tile (+pad)

    const int tid   = threadIdx.x;
    const int tileM = blockIdx.x * 128;
    const int tileN = blockIdx.y * 64;
    const int w     = tid >> 5;
    const int lane  = tid & 31;
    const int half  = lane >> 4;
    const int r16   = lane & 15;

    v8f acc[4] = {};

    for (int k0 = 0; k0 < KIN; k0 += 32) {
        __syncthreads();
        // --- cooperative A load: 128 rows x 32 cols, 16B chunks ---
        #pragma unroll
        for (int i = 0; i < 2; ++i) {
            int ch = tid * 2 + i;
            int r  = ch >> 2;
            int c8 = (ch & 3) * 8;
            uint4 v = *(const uint4*)(Xb + (size_t)(tileM + r) * KIN + k0 + c8);
            *(uint4*)&As[r][c8] = v;
        }
        // --- cooperative B load: 32(K) x 64(N), transposed into Bs[N][K] ---
        {
            int kk = tid >> 3;
            int c8 = (tid & 7) * 8;
            uint4 v = *(const uint4*)(Wb + (size_t)(k0 + kk) * NCOLS + tileN + c8);
            const bf16_t* pv = (const bf16_t*)&v;
            #pragma unroll
            for (int i = 0; i < 8; ++i) Bs[c8 + i][kk] = pv[i];
        }
        __syncthreads();

        // A fragment (16-bit A 16x32 layout): lo8 = K[half*8 .. +7],
        // hi8 = K[16 + half*8 .. +7], row M = lane%16.
        union { uint4 u[2]; v16bf v; } au;
        au.u[0] = *(const uint4*)&As[w * 16 + r16][half * 8];
        au.u[1] = *(const uint4*)&As[w * 16 + r16][16 + half * 8];

        #pragma unroll
        for (int nt = 0; nt < 4; ++nt) {
            // B fragment: lane -> col N = lane%16, lo8 = K[half*8 .. +7], hi8 = K[16 + half*8 .. +7].
            union { uint4 u[2]; v16bf v; } bu;
            bu.u[0] = *(const uint4*)&Bs[nt * 16 + r16][half * 8];
            bu.u[1] = *(const uint4*)&Bs[nt * 16 + r16][16 + half * 8];
            acc[nt] = __builtin_amdgcn_wmma_f32_16x16x32_bf16(
                false, au.v, false, bu.v, (short)0, acc[nt], false, false);
        }
    }

    // --- epilogue: scatter into Uperm[(m*2+dir)*H + h][4] ---
    #pragma unroll
    for (int nt = 0; nt < 4; ++nt) {
        int n   = tileN + nt * 16 + r16;
        int dir = n / (3 * H);
        int jj  = (n / H) % 3;
        int h   = n & (H - 1);
        #pragma unroll
        for (int j = 0; j < 8; ++j) {
            // C/D layout: VGPR j -> M = j (lanes 0-15), j+8 (lanes 16-31)
            int m = tileM + w * 16 + half * 8 + j;
            Uperm[(size_t)((m * 2 + dir) * H + h) * 4 + jj] = acc[nt][j];
        }
        if (jj == 0) {  // fill residual slot (k==3 layers: res = x itself)
            #pragma unroll
            for (int j = 0; j < 8; ++j) {
                int m = tileM + w * 16 + half * 8 + j;
                float res = (float)Xb[(size_t)m * KIN + dir * H + h];
                Uperm[(size_t)((m * 2 + dir) * H + h) * 4 + 3] = res;
            }
        }
    }
}

// ---------------------------------------------------------------------------
// SRU recurrence: 8192 channels (b, dir, h), 2048 sequential steps each.
// Backward direction walks l = L-1 .. 0 (flips never materialized).
// Software-pipelined: prefetch next step's float4 while computing current.
// Writes next-layer activation as bf16 into xout[m*1024 + dir*H + h].
// ---------------------------------------------------------------------------
__global__ __launch_bounds__(256)
void sru_scan_kernel(const float4* __restrict__ Uperm,
                     const float* __restrict__ wc,
                     const float* __restrict__ bias,
                     bf16_t* __restrict__ xout) {
    int tid = blockIdx.x * blockDim.x + threadIdx.x;  // 0 .. 8191
    int h   = tid & (H - 1);
    int dir = (tid >> 9) & 1;
    int b   = tid >> 10;

    float vf = wc[dir * H + h];
    float vr = wc[2 * H + dir * H + h];
    float bf = bias[dir * H + h];
    float br = bias[2 * H + dir * H + h];

    float c = 0.0f;
    int l    = dir ? (LSEQ - 1) : 0;
    int step = dir ? -1 : 1;

    float4 u = Uperm[((l * BDIM + b) * 2 + dir) * H + h];
    for (int t = 0; t < LSEQ; ++t) {
        int ln = l + step;
        float4 un = (t + 1 < LSEQ)
                        ? Uperm[((ln * BDIM + b) * 2 + dir) * H + h]
                        : u;
        float f = sigmoidf_(u.y + vf * c + bf);
        c = f * c + (1.0f - f) * u.x;
        float r = sigmoidf_(u.z + vr * c + br);
        float hh = r * c + (1.0f - r) * u.w;
        xout[(size_t)(l * BDIM + b) * KIN + dir * H + h] = (bf16_t)hh;
        u = un;
        l = ln;
    }
}

// ---------------------------------------------------------------------------
// Classifier: y[b,l,:] = x[l,b,:] @ Wcls(1024x3) + bcls.  One block per row.
// ---------------------------------------------------------------------------
__global__ __launch_bounds__(128)
void cls_kernel(const bf16_t* __restrict__ xin,
                const float* __restrict__ Wc,
                const float* __restrict__ bc,
                float* __restrict__ out) {
    __shared__ float red[3][128];
    int m   = blockIdx.x;
    int tid = threadIdx.x;
    float s0 = 0.f, s1 = 0.f, s2 = 0.f;
    #pragma unroll
    for (int i = 0; i < 8; ++i) {
        int n = i * 128 + tid;
        float xv = (float)xin[(size_t)m * KIN + n];
        s0 += xv * Wc[n * 3 + 0];
        s1 += xv * Wc[n * 3 + 1];
        s2 += xv * Wc[n * 3 + 2];
    }
    red[0][tid] = s0; red[1][tid] = s1; red[2][tid] = s2;
    __syncthreads();
    for (int s = 64; s > 0; s >>= 1) {
        if (tid < s) {
            red[0][tid] += red[0][tid + s];
            red[1][tid] += red[1][tid + s];
            red[2][tid] += red[2][tid + s];
        }
        __syncthreads();
    }
    if (tid == 0) {
        int b = m & 7, l = m >> 3;
        size_t o = (size_t)(b * LSEQ + l) * 3;
        out[o + 0] = red[0][0] + bc[0];
        out[o + 1] = red[1][0] + bc[1];
        out[o + 2] = red[2][0] + bc[2];
    }
}

// ---------------------------------------------------------------------------
// Launch sequence (all on `stream`; ws layout fully overwritten each call):
//   [Uperm f32: 256 MB][x bf16: 32 MB][W1..3 bf16: 18 MB]
// ---------------------------------------------------------------------------
extern "C" void kernel_launch(void* const* d_in, const int* in_sizes, int n_in,
                              void* d_out, int out_size, void* d_ws, size_t ws_size,
                              hipStream_t stream) {
    (void)in_sizes; (void)n_in; (void)out_size; (void)ws_size;

    const float* X    = (const float*)d_in[0];
    const float* W0   = (const float*)d_in[1];
    const float* wc0  = (const float*)d_in[2];
    const float* b0   = (const float*)d_in[3];
    const float* Wl[3]  = {(const float*)d_in[4], (const float*)d_in[7],  (const float*)d_in[10]};
    const float* wcl[3] = {(const float*)d_in[5], (const float*)d_in[8],  (const float*)d_in[11]};
    const float* bl[3]  = {(const float*)d_in[6], (const float*)d_in[9],  (const float*)d_in[12]};
    const float* Wcls = (const float*)d_in[13];
    const float* bcls = (const float*)d_in[14];

    char* ws = (char*)d_ws;
    float*  Uperm = (float*)ws;                                  // 16384*2*512*4 f32 = 268435456 B
    bf16_t* xbf   = (bf16_t*)(ws + 268435456);                   // 16384*1024 bf16 = 33554432 B
    bf16_t* Wb    = (bf16_t*)(ws + 268435456 + 33554432);        // 3 * 1024*3072 bf16

    const int wn = KIN * NCOLS;  // 3145728 elems per layer weight

    // Convert layer 1..3 weights to bf16.
    for (int i = 0; i < 3; ++i) {
        hipLaunchKernelGGL(f32_to_bf16_kernel, dim3(wn / 256), dim3(256), 0, stream,
                           Wl[i], Wb + (size_t)i * wn, wn);
    }

    // Layer 0 (k=4, n_in=1): dX + outer product straight into Uperm.
    hipLaunchKernelGGL(layer0_u_kernel, dim3(MROWS * 2 * H / 256), dim3(256), 0, stream,
                       X, W0, (float4*)Uperm);
    hipLaunchKernelGGL(sru_scan_kernel, dim3(2 * BDIM * H / 256), dim3(256), 0, stream,
                       (const float4*)Uperm, wc0, b0, xbf);

    // Layers 1..3 (k=3): WMMA GEMM + scan.
    for (int i = 0; i < 3; ++i) {
        hipLaunchKernelGGL(sru_gemm_kernel, dim3(MROWS / 128, NCOLS / 64), dim3(256), 0, stream,
                           xbf, Wb + (size_t)i * wn, Uperm);
        hipLaunchKernelGGL(sru_scan_kernel, dim3(2 * BDIM * H / 256), dim3(256), 0, stream,
                           (const float4*)Uperm, wcl[i], bl[i], xbf);
    }

    // Classifier.
    hipLaunchKernelGGL(cls_kernel, dim3(MROWS), dim3(128), 0, stream,
                       xbf, Wcls, bcls, (float*)d_out);
}
